// DiffusionDenoiser_19679540150753
// MI455X (gfx1250) — compile-verified, matrix operand order checked by reading, not run on hardware
//
#include <hip/hip_runtime.h>
#include <cstdint>

// Problem constants (match reference)
#define ND   192
#define ED   64
#define CD   128
#define HIDN 256
#define NLAY 6
#define BB   16
#define NNODE 512
#define EEDGE 16384
#define MAXZV 100

typedef __attribute__((ext_vector_type(16))) __bf16 v16bf;
typedef __attribute__((ext_vector_type(8)))  __bf16 v8bf;
typedef __attribute__((ext_vector_type(8)))  float  v8f;

__device__ __forceinline__ float silu_f(float x) { return x / (1.0f + __expf(-x)); }

__device__ __forceinline__ v8f wmma_bf16_acc(v16bf a, v16bf b, v8f c) {
  // D = A(16x32) * B(32x16) + C, fp32 accumulate
  return __builtin_amdgcn_wmma_f32_16x16x32_bf16(false, a, false, b, (short)0, c,
                                                 false, false);
}

// A-fragment load from LDS row-major [16 rows x ld cols] bf16.
// ISA 16-bit A 16x32 layout: lanes0-15 row=lane, K={0..7,16..23};
// lanes16-31 row=lane-16, K={8..15,24..31}.
__device__ __forceinline__ v16bf load_afrag(const __bf16* base, int ld, int kt) {
  int lane = threadIdx.x & 31;
  const __bf16* p = base + (lane & 15) * ld + kt * 32 + ((lane & 16) ? 8 : 0);
  v8bf lo = *(const v8bf*)p;
  v8bf hi = *(const v8bf*)(p + 16);
  v16bf r;
#pragma unroll
  for (int i = 0; i < 8; ++i) { r[i] = lo[i]; r[i + 8] = hi[i]; }
  return r;
}

// B-fragment load from pre-repacked weights: each (ktile,ntile) tile stores
// 32 lanes x 16 bf16 contiguously in the exact per-lane fragment order.
__device__ __forceinline__ v16bf load_bfrag(const __bf16* tile) {
  const __bf16* p = tile + (threadIdx.x & 31) * 16;
  v8bf lo = *(const v8bf*)p;
  v8bf hi = *(const v8bf*)(p + 8);
  v16bf r;
#pragma unroll
  for (int i = 0; i < 8; ++i) { r[i] = lo[i]; r[i + 8] = hi[i]; }
  return r;
}

// ---------------- weight repack: fp32 KxN row-major -> bf16 fragment order ---
__global__ void repack_weight_bf16(const float* __restrict__ src,
                                   __bf16* __restrict__ dst, int K, int N) {
  int idx = blockIdx.x * blockDim.x + threadIdx.x;
  int total = K * N;
  if (idx >= total) return;
  int j = idx & 15;
  int lane = (idx >> 4) & 31;
  int tile = idx >> 9;
  int ntiles = N >> 4;
  int ktile = tile / ntiles;
  int ntile = tile - ktile * ntiles;
  int n = ntile * 16 + (lane & 15);
  int klocal = ((j < 8) ? j : (j + 8)) + ((lane & 16) ? 8 : 0);
  int k = ktile * 32 + klocal;
  dst[idx] = (__bf16)src[k * N + n];
}

// ---------------- small embedding kernels (scalar fp32, negligible FLOPs) ----
__global__ void coord1_kernel(const float* __restrict__ x_t,
                              const float* __restrict__ w1,
                              const float* __restrict__ b1,
                              float* __restrict__ tmp) {
  int idx = blockIdx.x * blockDim.x + threadIdx.x;
  if (idx >= BB * NNODE * ND) return;
  int row = idx / ND, c = idx - row * ND;
  float x0 = x_t[row * 3 + 0];
  float x1 = x_t[row * 3 + 1];
  float x2 = x_t[row * 3 + 2] * 0.1f;  // enforce_2d z-scale
  float a = b1[c] + x0 * w1[c] + x1 * w1[ND + c] + x2 * w1[2 * ND + c];
  tmp[idx] = silu_f(a);
}

__global__ void coord2_kernel(const float* __restrict__ tmp,
                              const float* __restrict__ w2,
                              const float* __restrict__ b2,
                              const int* __restrict__ z,
                              const float* __restrict__ atom_table,
                              float* __restrict__ h0) {
  int idx = blockIdx.x * blockDim.x + threadIdx.x;
  if (idx >= BB * NNODE * ND) return;
  int row = idx / ND, c = idx - row * ND;
  float a = b2[c];
  const float* tr = tmp + row * ND;
  for (int k = 0; k < ND; ++k) a += tr[k] * w2[k * ND + c];
  int zz = z[row];
  if (zz < 0) zz = 0;
  if (zz > MAXZV) zz = MAXZV;
  h0[idx] = a + atom_table[zz * ND + c];
}

// edge embedding: RBF(33) -> 64 silu -> 64 ; 4 rows per 256-thread block
__global__ void edge_embed_kernel(const float* __restrict__ dist,
                                  const float* __restrict__ centers,
                                  const float* __restrict__ gamma,
                                  const float* __restrict__ w1,
                                  const float* __restrict__ b1,
                                  const float* __restrict__ w2,
                                  const float* __restrict__ b2,
                                  __bf16* __restrict__ e_bf) {
  __shared__ float s1[4][ED];
  int tid = threadIdx.x;
  int r = tid >> 6, c = tid & 63;
  int grow = blockIdx.x * 4 + r;  // grow == b*E + eidx (b-major flatten)
  float d = dist[grow];
  d = fminf(fmaxf(d, 0.0f), 5.0f);
  float g = gamma[0];
  float a = b1[c] + (d * 0.2f) * w1[32 * ED + c];
  for (int k = 0; k < 32; ++k) {
    float dd = d - centers[k];
    a += __expf(-g * dd * dd) * w1[k * ED + c];
  }
  s1[r][c] = silu_f(a);
  __syncthreads();
  float a2 = b2[c];
  for (int k = 0; k < ED; ++k) a2 += s1[r][k] * w2[k * ED + c];
  e_bf[(size_t)grow * ED + c] = (__bf16)a2;
}

// time + property conditioning, one block
__global__ void cond_kernel(const float* __restrict__ t, const float* __restrict__ y,
                            const float* __restrict__ tw1, const float* __restrict__ tb1,
                            const float* __restrict__ tw2, const float* __restrict__ tb2,
                            const float* __restrict__ pw1, const float* __restrict__ pb1,
                            const float* __restrict__ pw2, const float* __restrict__ pb2,
                            const float* __restrict__ pw3, const float* __restrict__ pb3,
                            __bf16* __restrict__ cond_bf) {
  __shared__ float temb0[BB][128];
  __shared__ float tmp1[BB][256];
  __shared__ float tmp2[BB][256];
  __shared__ float yb2[BB][256];
  int tid = threadIdx.x;
  const float lf = logf(10000.0f) / 63.0f;
  for (int i = tid; i < BB * 128; i += blockDim.x) {
    int b = i >> 7, c = i & 127;
    int j = c & 63;
    float ang = t[b] * __expf(-lf * (float)j);
    temb0[b][c] = (c < 64) ? __sinf(ang) : __cosf(ang);
  }
  __syncthreads();
  for (int i = tid; i < BB * 128; i += blockDim.x) {  // temb1 = silu(temb0@tw1+tb1)
    int b = i >> 7, c = i & 127;
    float a = tb1[c];
    for (int k = 0; k < 128; ++k) a += temb0[b][k] * tw1[k * 128 + c];
    tmp1[b][c] = silu_f(a);
  }
  __syncthreads();
  for (int i = tid; i < BB * 128; i += blockDim.x) {  // temb2 = temb1@tw2+tb2
    int b = i >> 7, c = i & 127;
    float a = tb2[c];
    for (int k = 0; k < 128; ++k) a += tmp1[b][k] * tw2[k * 128 + c];
    tmp2[b][c] = a;
  }
  __syncthreads();
  for (int i = tid; i < BB * 256; i += blockDim.x) {  // yemb1 = silu(y@pw1+pb1)
    int b = i >> 8, c = i & 255;
    float a = pb1[c] + y[b * 3] * pw1[c] + y[b * 3 + 1] * pw1[256 + c] +
              y[b * 3 + 2] * pw1[512 + c];
    tmp1[b][c] = silu_f(a);
  }
  __syncthreads();
  for (int i = tid; i < BB * 256; i += blockDim.x) {  // yemb2 = silu(yemb1@pw2+pb2)
    int b = i >> 8, c = i & 255;
    float a = pb2[c];
    for (int k = 0; k < 256; ++k) a += tmp1[b][k] * pw2[k * 256 + c];
    yb2[b][c] = silu_f(a);
  }
  __syncthreads();
  for (int i = tid; i < BB * 128; i += blockDim.x) {  // cond = temb2 + yemb2@pw3+pb3
    int b = i >> 7, c = i & 127;
    float a = pb3[c];
    for (int k = 0; k < 256; ++k) a += yb2[b][k] * pw3[k * 128 + c];
    a += tmp2[b][c];
    cond_bf[i] = (__bf16)a;
  }
}

// ---------------- fused edge message kernel (the hot loop) ------------------
// 64 edge rows per block (4 waves x 16-row WMMA tiles), 128 threads.
// LDS: m_in 64x384 bf16 (48KB) + hidden 64x256 bf16 (32KB) = 80KB dynamic.
__global__ void edge_msg_kernel(const float* __restrict__ h,
                                const __bf16* __restrict__ e_bf,
                                const __bf16* __restrict__ cond_bf,
                                const int* __restrict__ edge_index,
                                const float* __restrict__ edge_mask,
                                const __bf16* __restrict__ w1,
                                const float* __restrict__ b1,
                                const __bf16* __restrict__ w2,
                                const float* __restrict__ b2,
                                float* __restrict__ agg) {
  extern __shared__ char smem[];
  __bf16* s_in = (__bf16*)smem;                       // 64 x 384
  __bf16* s_hid = (__bf16*)(smem + 64 * 384 * 2);     // 64 x 256
  const int row0 = blockIdx.x * 64;
  const int tid = threadIdx.x;

  // gather: m_in = [h[b,src,:192] | e[b,eidx,:64] | cond[b,:128]] as bf16
  for (int i = tid; i < 64 * 384; i += blockDim.x) {
    int r = i / 384, c = i - r * 384;
    int grow = row0 + r;
    int b = grow >> 14;        // E = 16384
    int eidx = grow & (EEDGE - 1);
    __bf16 v;
    if (c < ND) {
      int src = edge_index[eidx];
      v = (__bf16)h[((size_t)b * NNODE + src) * ND + c];
    } else if (c < ND + ED) {
      v = e_bf[((size_t)b * EEDGE + eidx) * ED + (c - ND)];
    } else {
      v = cond_bf[b * CD + (c - ND - ED)];
    }
    s_in[i] = v;
  }
  __syncthreads();

  const int wave = tid >> 5;
  const int lane = tid & 31;
  const __bf16* a1 = s_in + wave * 16 * 384;

  // msg1: (16x384) @ (384x256), silu -> LDS bf16
  for (int nt = 0; nt < 16; ++nt) {
    v8f acc = {};
#pragma unroll
    for (int kt = 0; kt < 12; ++kt) {
      v16bf a = load_afrag(a1, 384, kt);
      v16bf bm = load_bfrag(w1 + (size_t)(kt * 16 + nt) * 512);
      acc = wmma_bf16_acc(a, bm, acc);
    }
    int n = nt * 16 + (lane & 15);
    float bias = b1[n];
#pragma unroll
    for (int r = 0; r < 8; ++r) {
      int m = r + ((lane & 16) ? 8 : 0);
      float v = silu_f(acc[r] + bias);
      s_hid[(wave * 16 + m) * 256 + n] = (__bf16)v;
    }
  }
  __syncthreads();

  // msg2: (16x256) @ (256x256), silu * edge_mask, scatter-add to agg
  const __bf16* a2 = s_hid + wave * 16 * 256;
  const int gbase = row0 + wave * 16;
  for (int nt = 0; nt < 16; ++nt) {
    v8f acc = {};
#pragma unroll
    for (int kt = 0; kt < 8; ++kt) {
      v16bf a = load_afrag(a2, 256, kt);
      v16bf bm = load_bfrag(w2 + (size_t)(kt * 16 + nt) * 512);
      acc = wmma_bf16_acc(a, bm, acc);
    }
    int n = nt * 16 + (lane & 15);
    float bias = b2[n];
#pragma unroll
    for (int r = 0; r < 8; ++r) {
      int m = r + ((lane & 16) ? 8 : 0);
      int grow = gbase + m;
      int b = grow >> 14, eidx = grow & (EEDGE - 1);
      float v = silu_f(acc[r] + bias) * edge_mask[grow];
      int dst = edge_index[EEDGE + eidx];
      atomicAdd(&agg[((size_t)b * NNODE + dst) * HIDN + n], v);
    }
  }
}

// ---------------- fused node update: GEMMs + residual + LayerNorm -----------
// 64 node rows per block, 128 threads.
// LDS: u_in 64x576 bf16 (72KB) + hid 64x256 bf16 (32KB) + r 64x192 f32 (48KB)
__global__ void node_upd_kernel(const float* __restrict__ h,
                                const float* __restrict__ agg,
                                const __bf16* __restrict__ cond_bf,
                                const unsigned char* __restrict__ mask,
                                const __bf16* __restrict__ w1,
                                const float* __restrict__ b1,
                                const __bf16* __restrict__ w2,
                                const float* __restrict__ b2,
                                const float* __restrict__ ln_g,
                                const float* __restrict__ ln_b,
                                float* __restrict__ h_out) {
  extern __shared__ char smem[];
  __bf16* s_in = (__bf16*)smem;                                  // 64 x 576
  __bf16* s_hid = (__bf16*)(smem + 64 * 576 * 2);                // 64 x 256
  float* s_r = (float*)(smem + 64 * 576 * 2 + 64 * 256 * 2);     // 64 x 192
  const int row0 = blockIdx.x * 64;
  const int tid = threadIdx.x;

  for (int i = tid; i < 64 * 576; i += blockDim.x) {
    int r = i / 576, c = i - r * 576;
    int grow = row0 + r;
    float v;
    if (c < ND) v = h[(size_t)grow * ND + c];
    else if (c < ND + HIDN) v = agg[(size_t)grow * HIDN + (c - ND)];
    else v = (float)cond_bf[(grow >> 9) * CD + (c - ND - HIDN)];
    s_in[i] = (__bf16)v;
  }
  __syncthreads();

  const int wave = tid >> 5, lane = tid & 31;
  const __bf16* a1 = s_in + wave * 16 * 576;
  for (int nt = 0; nt < 16; ++nt) {  // (16x576)@(576x256) silu
    v8f acc = {};
#pragma unroll
    for (int kt = 0; kt < 18; ++kt) {
      v16bf a = load_afrag(a1, 576, kt);
      v16bf bm = load_bfrag(w1 + (size_t)(kt * 16 + nt) * 512);
      acc = wmma_bf16_acc(a, bm, acc);
    }
    int n = nt * 16 + (lane & 15);
    float bias = b1[n];
#pragma unroll
    for (int r = 0; r < 8; ++r) {
      int m = r + ((lane & 16) ? 8 : 0);
      s_hid[(wave * 16 + m) * 256 + n] = (__bf16)silu_f(acc[r] + bias);
    }
  }
  __syncthreads();

  const __bf16* a2 = s_hid + wave * 16 * 256;
  for (int nt = 0; nt < 12; ++nt) {  // (16x256)@(256x192) + residual
    v8f acc = {};
#pragma unroll
    for (int kt = 0; kt < 8; ++kt) {
      v16bf a = load_afrag(a2, 256, kt);
      v16bf bm = load_bfrag(w2 + (size_t)(kt * 12 + nt) * 512);
      acc = wmma_bf16_acc(a, bm, acc);
    }
    int n = nt * 16 + (lane & 15);
    float bias = b2[n];
#pragma unroll
    for (int r = 0; r < 8; ++r) {
      int m = r + ((lane & 16) ? 8 : 0);
      int grow = row0 + wave * 16 + m;
      s_r[(wave * 16 + m) * ND + n] = acc[r] + bias + h[(size_t)grow * ND + n];
    }
  }
  __syncthreads();

  if (lane < 16) {  // LayerNorm + mask, one lane per row
    int lrow = wave * 16 + lane;
    int grow = row0 + lrow;
    const float* rr = s_r + lrow * ND;
    float mu = 0.0f;
    for (int i = 0; i < ND; ++i) mu += rr[i];
    mu *= (1.0f / ND);
    float var = 0.0f;
    for (int i = 0; i < ND; ++i) { float d = rr[i] - mu; var += d * d; }
    var *= (1.0f / ND);
    float inv = rsqrtf(var + 1e-5f);
    unsigned char mk = mask[grow];
    for (int i = 0; i < ND; ++i) {
      float o = (rr[i] - mu) * inv * ln_g[i] + ln_b[i];
      h_out[(size_t)grow * ND + i] = mk ? o : h[(size_t)grow * ND + i];
    }
  }
}

// ---------------- output head: 320->256 silu ->256 silu ->3 ------------------
__global__ void out_kernel(const float* __restrict__ h,
                           const __bf16* __restrict__ cond_bf,
                           const unsigned char* __restrict__ mask,
                           const __bf16* __restrict__ w1, const float* __restrict__ b1,
                           const __bf16* __restrict__ w2, const float* __restrict__ b2,
                           const float* __restrict__ w3, const float* __restrict__ b3,
                           float* __restrict__ out) {
  extern __shared__ char smem[];
  __bf16* s_in = (__bf16*)smem;                                  // 64 x 320
  __bf16* s_h1 = (__bf16*)(smem + 64 * 320 * 2);                 // 64 x 256
  __bf16* s_h2 = (__bf16*)(smem + 64 * 320 * 2 + 64 * 256 * 2);  // 64 x 256
  const int row0 = blockIdx.x * 64;
  const int tid = threadIdx.x;
  for (int i = tid; i < 64 * 320; i += blockDim.x) {
    int r = i / 320, c = i - r * 320;
    int grow = row0 + r;
    float v = (c < ND) ? h[(size_t)grow * ND + c]
                       : (float)cond_bf[(grow >> 9) * CD + (c - ND)];
    s_in[i] = (__bf16)v;
  }
  __syncthreads();
  const int wave = tid >> 5, lane = tid & 31;
  const __bf16* a1 = s_in + wave * 16 * 320;
  for (int nt = 0; nt < 16; ++nt) {
    v8f acc = {};
#pragma unroll
    for (int kt = 0; kt < 10; ++kt) {
      v16bf a = load_afrag(a1, 320, kt);
      v16bf bm = load_bfrag(w1 + (size_t)(kt * 16 + nt) * 512);
      acc = wmma_bf16_acc(a, bm, acc);
    }
    int n = nt * 16 + (lane & 15);
    float bias = b1[n];
#pragma unroll
    for (int r = 0; r < 8; ++r) {
      int m = r + ((lane & 16) ? 8 : 0);
      s_h1[(wave * 16 + m) * 256 + n] = (__bf16)silu_f(acc[r] + bias);
    }
  }
  __syncthreads();
  const __bf16* a2 = s_h1 + wave * 16 * 256;
  for (int nt = 0; nt < 16; ++nt) {
    v8f acc = {};
#pragma unroll
    for (int kt = 0; kt < 8; ++kt) {
      v16bf a = load_afrag(a2, 256, kt);
      v16bf bm = load_bfrag(w2 + (size_t)(kt * 16 + nt) * 512);
      acc = wmma_bf16_acc(a, bm, acc);
    }
    int n = nt * 16 + (lane & 15);
    float bias = b2[n];
#pragma unroll
    for (int r = 0; r < 8; ++r) {
      int m = r + ((lane & 16) ? 8 : 0);
      s_h2[(wave * 16 + m) * 256 + n] = (__bf16)silu_f(acc[r] + bias);
    }
  }
  __syncthreads();
  if (lane < 16) {
    int lrow = wave * 16 + lane;
    int grow = row0 + lrow;
    unsigned char mk = mask[grow];
    for (int o = 0; o < 3; ++o) {
      float acc = b3[o];
      for (int k = 0; k < 256; ++k) acc += (float)s_h2[lrow * 256 + k] * w3[k * 3 + o];
      out[(size_t)grow * 3 + o] = mk ? acc : 0.0f;
    }
  }
}

// ---------------- host launch ------------------------------------------------
extern "C" void kernel_launch(void* const* d_in, const int* in_sizes, int n_in,
                              void* d_out, int out_size, void* d_ws, size_t ws_size,
                              hipStream_t stream) {
  (void)in_sizes; (void)n_in; (void)out_size; (void)ws_size;
  // inputs in setup_inputs() order
  const int*   z          = (const int*)d_in[0];
  const float* x_t        = (const float*)d_in[1];
  const int*   edge_index = (const int*)d_in[2];
  const float* dist       = (const float*)d_in[3];
  const float* t_in       = (const float*)d_in[4];
  const float* y_in       = (const float*)d_in[5];
  const unsigned char* mask = (const unsigned char*)d_in[6];
  const float* edge_mask  = (const float*)d_in[7];
  const float* atom_table = (const float*)d_in[8];
  const float* coord_w1 = (const float*)d_in[9];  const float* coord_b1 = (const float*)d_in[10];
  const float* coord_w2 = (const float*)d_in[11]; const float* coord_b2 = (const float*)d_in[12];
  const float* edge_centers = (const float*)d_in[13]; const float* edge_gamma = (const float*)d_in[14];
  const float* edge_w1 = (const float*)d_in[15]; const float* edge_b1 = (const float*)d_in[16];
  const float* edge_w2 = (const float*)d_in[17]; const float* edge_b2 = (const float*)d_in[18];
  const float* time_w1 = (const float*)d_in[19]; const float* time_b1 = (const float*)d_in[20];
  const float* time_w2 = (const float*)d_in[21]; const float* time_b2 = (const float*)d_in[22];
  const float* prop_w1 = (const float*)d_in[23]; const float* prop_b1 = (const float*)d_in[24];
  const float* prop_w2 = (const float*)d_in[25]; const float* prop_b2 = (const float*)d_in[26];
  const float* prop_w3 = (const float*)d_in[27]; const float* prop_b3 = (const float*)d_in[28];
  const float* msg_w1 = (const float*)d_in[29]; const float* msg_b1 = (const float*)d_in[30];
  const float* msg_w2 = (const float*)d_in[31]; const float* msg_b2 = (const float*)d_in[32];
  const float* upd_w1 = (const float*)d_in[33]; const float* upd_b1 = (const float*)d_in[34];
  const float* upd_w2 = (const float*)d_in[35]; const float* upd_b2 = (const float*)d_in[36];
  const float* ln_g = (const float*)d_in[37]; const float* ln_b = (const float*)d_in[38];
  const float* out_w1 = (const float*)d_in[39]; const float* out_b1 = (const float*)d_in[40];
  const float* out_w2 = (const float*)d_in[41]; const float* out_b2 = (const float*)d_in[42];
  const float* out_w3 = (const float*)d_in[43]; const float* out_b3 = (const float*)d_in[44];

  // workspace carve-out (256B aligned)
  size_t off = 0;
  auto carve = [&](size_t bytes) -> void* {
    void* p = (char*)d_ws + off;
    off = (off + bytes + 255) & ~(size_t)255;
    return p;
  };
  float*  h_a       = (float*)carve((size_t)BB * NNODE * ND * 4);
  float*  h_b       = (float*)carve((size_t)BB * NNODE * ND * 4);
  float*  agg       = (float*)carve((size_t)BB * NNODE * HIDN * 4);
  float*  coord_tmp = (float*)carve((size_t)BB * NNODE * ND * 4);
  __bf16* e_bf      = (__bf16*)carve((size_t)BB * EEDGE * ED * 2);
  __bf16* cond_bf   = (__bf16*)carve((size_t)BB * CD * 2);
  __bf16* msgw1bf   = (__bf16*)carve((size_t)NLAY * 384 * 256 * 2);
  __bf16* msgw2bf   = (__bf16*)carve((size_t)NLAY * 256 * 256 * 2);
  __bf16* updw1bf   = (__bf16*)carve((size_t)NLAY * 576 * 256 * 2);
  __bf16* updw2bf   = (__bf16*)carve((size_t)NLAY * 256 * 192 * 2);
  __bf16* outw1bf   = (__bf16*)carve((size_t)320 * 256 * 2);
  __bf16* outw2bf   = (__bf16*)carve((size_t)256 * 256 * 2);

  auto repack = [&](const float* src, __bf16* dst, int K, int N) {
    int tot = K * N;
    repack_weight_bf16<<<(tot + 255) / 256, 256, 0, stream>>>(src, dst, K, N);
  };
  for (int l = 0; l < NLAY; ++l) {
    repack(msg_w1 + (size_t)l * 384 * 256, msgw1bf + (size_t)l * 384 * 256, 384, 256);
    repack(msg_w2 + (size_t)l * 256 * 256, msgw2bf + (size_t)l * 256 * 256, 256, 256);
    repack(upd_w1 + (size_t)l * 576 * 256, updw1bf + (size_t)l * 576 * 256, 576, 256);
    repack(upd_w2 + (size_t)l * 256 * 192, updw2bf + (size_t)l * 256 * 192, 256, 192);
  }
  repack(out_w1, outw1bf, 320, 256);
  repack(out_w2, outw2bf, 256, 256);

  // embeddings
  cond_kernel<<<1, 256, 0, stream>>>(t_in, y_in, time_w1, time_b1, time_w2, time_b2,
                                     prop_w1, prop_b1, prop_w2, prop_b2, prop_w3,
                                     prop_b3, cond_bf);
  {
    int tot = BB * NNODE * ND;
    coord1_kernel<<<(tot + 255) / 256, 256, 0, stream>>>(x_t, coord_w1, coord_b1, coord_tmp);
    coord2_kernel<<<(tot + 255) / 256, 256, 0, stream>>>(coord_tmp, coord_w2, coord_b2,
                                                         z, atom_table, h_a);
  }
  edge_embed_kernel<<<(BB * EEDGE) / 4, 256, 0, stream>>>(dist, edge_centers, edge_gamma,
                                                          edge_w1, edge_b1, edge_w2,
                                                          edge_b2, e_bf);

  // message-passing layers
  const size_t smem_edge = 64 * 384 * 2 + 64 * 256 * 2;                    // 80 KB
  const size_t smem_node = 64 * 576 * 2 + 64 * 256 * 2 + 64 * 192 * 4;     // 152 KB
  const size_t smem_out  = 64 * 320 * 2 + 64 * 256 * 2 + 64 * 256 * 2;     // 104 KB
  float* h_cur = h_a;
  float* h_nxt = h_b;
  for (int l = 0; l < NLAY; ++l) {
    hipMemsetAsync(agg, 0, (size_t)BB * NNODE * HIDN * 4, stream);
    edge_msg_kernel<<<(BB * EEDGE) / 64, 128, smem_edge, stream>>>(
        h_cur, e_bf, cond_bf, edge_index, edge_mask,
        msgw1bf + (size_t)l * 384 * 256, msg_b1 + l * 256,
        msgw2bf + (size_t)l * 256 * 256, msg_b2 + l * 256, agg);
    node_upd_kernel<<<(BB * NNODE) / 64, 128, smem_node, stream>>>(
        h_cur, agg, cond_bf, mask,
        updw1bf + (size_t)l * 576 * 256, upd_b1 + l * 256,
        updw2bf + (size_t)l * 256 * 192, upd_b2 + l * 192,
        ln_g + l * ND, ln_b + l * ND, h_nxt);
    float* tmp = h_cur; h_cur = h_nxt; h_nxt = tmp;
  }

  out_kernel<<<(BB * NNODE) / 64, 128, smem_out, stream>>>(
      h_cur, cond_bf, mask, outw1bf, out_b1, outw2bf, out_b2, out_w3, out_b3,
      (float*)d_out);
}
